// MemoryEfficientEdgeAttention_43499428774425
// MI455X (gfx1250) — compile-verified, matrix operand order checked by reading, not run on hardware
//
#include <hip/hip_runtime.h>

typedef __attribute__((ext_vector_type(16))) _Float16 v16h;
typedef __attribute__((ext_vector_type(8)))  _Float16 v8h;
typedef __attribute__((ext_vector_type(8)))  float    v8f;
typedef __attribute__((ext_vector_type(4)))  unsigned int v4u;
typedef __attribute__((ext_vector_type(8)))  int      v8i;
typedef __attribute__((ext_vector_type(4)))  int      v4i;

#define E_N      8192
#define KNB      16
#define HIDDEN   128
#define HEADS    8
#define HEAD_DIM 16
#define NRAD     64
#define CUTOFF   10.0f
#define LN_EPS   1e-5f
#define MLP_IN   97

// LDS row strides (halves) chosen to stagger banks for 16-row b128 reads.
#define ACT_LD   136   // per-wave activation tile rows
#define SKV_LD   136   // k/v f16 rows
#define SRB_LD   72    // rbf rows
#define W128_LD  136   // LDS-resident Wt tiles (K=128)
#define WG_LD    40    // LDS-resident Wg1t tiles (K=32 padded)

// ---------------------------------------------------------------------------
// WMMA fragment helpers (gfx1250, wave32), per ISA 7.12.2.
// A (16x32 f16): lane holds row m=lane&15; halves are TWO CONTIGUOUS 8-half
//   runs at k = ko..ko+7 and 16+ko..16+ko+7 where ko = (lane>=16)?8:0.
// B fragments are A fragments taken on the transposed weight (row-major
//   [N][K]): lane n holds column n of W with the same K striping.
// C/D (16x16 f32): lane owns column n=lane&15, rows rbase..rbase+7,
//   rbase=(lane<16)?0:8.
// ---------------------------------------------------------------------------
__device__ __forceinline__ v16h combine8(v8h lo, v8h hi) {
  union { v16h v; v8h h[2]; } u;
  u.h[0] = lo; u.h[1] = hi;
  return u.v;
}

__device__ __forceinline__ v16h load_frag_rm(const _Float16* base, int ld) {
  const int lane = threadIdx.x & 31;
  const int r = lane & 15;
  const int ko = (lane >> 4) << 3;
  const _Float16* p = base + r * ld + ko;
  return combine8(*(const v8h*)p, *(const v8h*)(p + 16));
}

__device__ __forceinline__ v8f wmma_f16(v16h a, v16h b, v8f c) {
  return __builtin_amdgcn_wmma_f32_16x16x32_f16(false, a, false, b, (short)0, c,
                                                false, false);
}

__device__ __forceinline__ float silu_f(float x) {
  return x / (1.0f + __expf(-x));
}

// ---------------------------------------------------------------------------
// Tensor Data Mover: 1-D tile of `nelem` f16 elements, global -> LDS, with
// hardware padding producing a strided LDS layout (ISA 08 §8.3/8.4):
//   pad_interval code v: pad after every 2^(v+3) bytes
//   pad_amount  code a: insert (a+1) DWORDs of padding
// D# group0: count=1 | lds_addr | global_addr | type=2
// D# group1: data_size=2B, pad fields, tensor_dim0=nelem, tensor_dim1=1,
//            tile_dim0=nelem (1-D), stride0=nelem.  Groups 2/3 zero (<=2-D).
// This toolchain's builtin takes 6 operands (g0, g1, g2, g3, g4, cpol).
// ---------------------------------------------------------------------------
__device__ __forceinline__ void tdm_load_1d_padded(const void* gsrc,
                                                   unsigned lds_byte_off,
                                                   unsigned nelem,
                                                   unsigned pad_interval_code,
                                                   unsigned pad_amount_code) {
  const unsigned long long ga = (unsigned long long)gsrc;
  v4u g0;
  g0[0] = 1u;                                             // count=1 (valid)
  g0[1] = lds_byte_off;                                   // lds_addr
  g0[2] = (unsigned)(ga & 0xFFFFFFFFull);                 // global_addr[31:0]
  g0[3] = (unsigned)((ga >> 32) & 0x1FFFFFFull) | (2u << 30);  // [56:32]|type=2
  v8i g1;
  g1[0] = (int)((1u << 16)                                // data_size = 2B
              | (1u << 20)                                // pad_enable
              | (pad_interval_code << 22)
              | (pad_amount_code << 25));
  g1[1] = (int)((nelem & 0xFFFFu) << 16);                 // tensor_dim0[15:0]
  g1[2] = (int)(((nelem >> 16) & 0xFFFFu) | (1u << 16));  // dim0[31:16]|dim1=1
  g1[3] = (int)((nelem & 0xFFFFu) << 16);                 // tile_dim0
  g1[4] = 0;                                              // tile_dim1/2 unused
  g1[5] = (int)nelem;                                     // dim0_stride[31:0]
  g1[6] = 0;
  g1[7] = 0;
  const v4i z4 = {};
  const v8i z8 = {};
  __builtin_amdgcn_tensor_load_to_lds(g0, g1, z4, z4, z8, 0);
}

// ---------------------------------------------------------------------------
// Kernel 0: build f16 TRANSPOSED weights ([N][K] row-major) + f16 ef.
// W1 K-padded 97->128, Wg1 K-padded 16->32.
// ---------------------------------------------------------------------------
__global__ void prep_kernel(const float* __restrict__ ef, const float* __restrict__ Wq,
                            const float* __restrict__ Wk, const float* __restrict__ Wv,
                            const float* __restrict__ W1, const float* __restrict__ W2,
                            const float* __restrict__ Wg1, const float* __restrict__ Wo,
                            _Float16* __restrict__ ef16, _Float16* __restrict__ WqT,
                            _Float16* __restrict__ WkT, _Float16* __restrict__ WvT,
                            _Float16* __restrict__ W1T, _Float16* __restrict__ W2T,
                            _Float16* __restrict__ Wg1T, _Float16* __restrict__ WoT) {
  const int i = blockIdx.x * blockDim.x + threadIdx.x;
  const int stride = gridDim.x * blockDim.x;
  for (int t = i; t < E_N * HIDDEN; t += stride) ef16[t] = (_Float16)ef[t];
  for (int t = i; t < HIDDEN * HIDDEN; t += stride) {
    const int n = t >> 7, k = t & 127;
    WqT[t] = (_Float16)Wq[k * HIDDEN + n];
    WkT[t] = (_Float16)Wk[k * HIDDEN + n];
    WvT[t] = (_Float16)Wv[k * HIDDEN + n];
    W2T[t] = (_Float16)W2[k * HIDDEN + n];
    WoT[t] = (_Float16)Wo[k * HIDDEN + n];
    W1T[t] = (k < MLP_IN) ? (_Float16)W1[k * HIDDEN + n] : (_Float16)0.0f;
  }
  for (int t = i; t < HIDDEN * 32; t += stride) {
    const int n = t >> 5, k = t & 31;
    Wg1T[t] = (k < HEAD_DIM) ? (_Float16)Wg1[k * HIDDEN + n] : (_Float16)0.0f;
  }
}

// ---------------------------------------------------------------------------
// Kernel 1: q/k/v = ef @ W.  One wave per 16-row tile; A loaded once,
// all 8 N-tiles computed (32 WMMA / wave).  grid (E/64, 3), block 128.
// ---------------------------------------------------------------------------
__global__ __launch_bounds__(128) void qkv_kernel(
    const _Float16* __restrict__ ef16, const _Float16* __restrict__ WqT,
    const _Float16* __restrict__ WkT, const _Float16* __restrict__ WvT,
    float* __restrict__ qf, float* __restrict__ kf, float* __restrict__ vf) {
  const int wave = threadIdx.x >> 5;
  const int tile = blockIdx.x * 4 + wave;
  const int mat = blockIdx.y;
  const _Float16* WT = (mat == 0) ? WqT : (mat == 1) ? WkT : WvT;
  float* out = (mat == 0) ? qf : (mat == 1) ? kf : vf;
  v16h a[4];
#pragma unroll
  for (int f = 0; f < 4; ++f)
    a[f] = load_frag_rm(ef16 + (size_t)tile * 16 * HIDDEN + f * 32, HIDDEN);
  const int lane = threadIdx.x & 31;
  const int n = lane & 15, rbase = (lane < 16) ? 0 : 8;
#pragma unroll
  for (int nt = 0; nt < 8; ++nt) {
    v8f acc = {};
#pragma unroll
    for (int f = 0; f < 4; ++f)
      acc = wmma_f16(a[f], load_frag_rm(WT + (nt * 16) * HIDDEN + f * 32, HIDDEN), acc);
#pragma unroll
    for (int r = 0; r < 8; ++r)
      out[(size_t)(tile * 16 + rbase + r) * HIDDEN + nt * 16 + n] = acc[r];
  }
}

// ---------------------------------------------------------------------------
// Kernel 2: one block per source edge (segment of exactly 16 pairs since
// src = repeat(arange(E),16)).  128 threads = 4 waves; wave w -> heads 2w,2w+1.
// ---------------------------------------------------------------------------
constexpr int K2_HALFS = 128 * W128_LD      // sW1T
                       + 128 * W128_LD      // sW2T
                       + 128 * WG_LD        // sWg1T
                       + 4 * 16 * ACT_LD    // sact
                       + 16 * SKV_LD        // skh
                       + 16 * SKV_LD        // svh
                       + 128                // sqh
                       + 16 * SRB_LD;       // srb
constexpr int K2_FLOATS = 16 * 128          // svf
                        + 128               // sqrow
                        + 128 * 8           // sW3
                        + 128 + 128 + 128 + 128  // sb1,sb2,sbg1,sWg2
                        + 8                 // sb3
                        + 16 * 8            // sdot
                        + 16 * 3            // scd
                        + 4 * 16 + 4 * 16   // sattn, sgate
                        + 4 + 1;            // scu, sbg2
constexpr size_t K2_LDS_BYTES = (size_t)K2_HALFS * 2 + (size_t)K2_FLOATS * 4 + 16 * 4;
// LDS byte offsets of the TDM-staged weight tiles (dynamic LDS starts at 0).
constexpr unsigned SOFF_W1 = 0;
constexpr unsigned SOFF_W2 = 128 * W128_LD * 2;
constexpr unsigned SOFF_WG = 2 * 128 * W128_LD * 2;

// One 8-half run of the X matrix row m for head h; run start c is a multiple
// of 8 so it never crosses a region boundary (16/32/96 are 8-aligned).
__device__ __forceinline__ v8h x_run(int c, int h, int m,
                                     const _Float16* sqh, const _Float16* skh,
                                     const _Float16* srb, const float* sdot) {
  if (c < 16)  return *(const v8h*)(sqh + h * HEAD_DIM + c);
  if (c < 32)  return *(const v8h*)(skh + m * SKV_LD + h * HEAD_DIM + (c - 16));
  if (c < 96)  return *(const v8h*)(srb + m * SRB_LD + (c - 32));
  v8h z = {};
  if (c == 96) z[0] = (_Float16)sdot[m * 8 + h];
  return z;
}

__global__ __launch_bounds__(128) void edge_attn_kernel(
    const float* __restrict__ qf, const float* __restrict__ kf,
    const float* __restrict__ vf, const float* __restrict__ coords,
    const int* __restrict__ pairs, const _Float16* __restrict__ W1T,
    const _Float16* __restrict__ W2T, const _Float16* __restrict__ Wg1T,
    const float* __restrict__ b1, const float* __restrict__ b2,
    const float* __restrict__ W3, const float* __restrict__ b3,
    const float* __restrict__ bg1, const float* __restrict__ Wg2,
    const float* __restrict__ bg2, _Float16* __restrict__ feats16,
    float* __restrict__ coords_out) {
  extern __shared__ __align__(16) char smem_raw[];
  _Float16* sW1T  = (_Float16*)smem_raw;
  _Float16* sW2T  = sW1T + 128 * W128_LD;
  _Float16* sWg1T = sW2T + 128 * W128_LD;
  _Float16* sact  = sWg1T + 128 * WG_LD;
  _Float16* skh   = sact + 4 * 16 * ACT_LD;
  _Float16* svh   = skh + 16 * SKV_LD;
  _Float16* sqh   = svh + 16 * SKV_LD;
  _Float16* srb   = sqh + 128;
  float* svf   = (float*)(srb + 16 * SRB_LD);
  float* sqrow = svf + 16 * 128;
  float* sW3   = sqrow + 128;
  float* sb1   = sW3 + 128 * 8;
  float* sb2   = sb1 + 128;
  float* sbg1  = sb2 + 128;
  float* sWg2  = sbg1 + 128;
  float* sb3   = sWg2 + 128;
  float* sdot  = sb3 + 8;
  float* scd   = sdot + 16 * 8;
  float* sattn = scd + 16 * 3;
  float* sgate = sattn + 4 * 16;
  float* scu   = sgate + 4 * 16;
  float* sbg2v = scu + 4;
  int* sdst    = (int*)(sbg2v + 1);

  const int e = blockIdx.x;
  const int tid = threadIdx.x;
  const int wave = tid >> 5;
  const int lane = tid & 31;

  // ---- Phase A: staging -------------------------------------------------
  // Weight tiles via the Tensor Data Mover; HW padding (4 DWORDs after every
  // 64/16 DWORDs) reproduces the W128_LD=136 / WG_LD=40 bank-staggered rows.
  if (wave == 0) {
    tdm_load_1d_padded(W1T, SOFF_W1, 128 * 128, /*256B*/ 5, /*4dw*/ 3);
    tdm_load_1d_padded(W2T, SOFF_W2, 128 * 128, /*256B*/ 5, /*4dw*/ 3);
    tdm_load_1d_padded(Wg1T, SOFF_WG, 128 * 32, /*64B*/ 3, /*4dw*/ 3);
    __builtin_amdgcn_s_wait_tensorcnt(0);
  }
  {  // small f32 tables
    const float4* w34 = (const float4*)W3;
    float4* sw34 = (float4*)sW3;
    for (int t = tid; t < 128 * 8 / 4; t += 128) sw34[t] = w34[t];
  }
  if (tid < 128) {
    sb1[tid] = b1[tid]; sb2[tid] = b2[tid];
    sbg1[tid] = bg1[tid]; sWg2[tid] = Wg2[tid];
    const float qv = qf[(size_t)e * HIDDEN + tid];
    sqrow[tid] = qv; sqh[tid] = (_Float16)qv;
  }
  if (tid < 8) sb3[tid] = b3[tid];
  if (tid == 0) *sbg2v = bg2[0];
  if (tid < 4) scu[tid] = 0.0f;
  if (tid < 16) sdst[tid] = pairs[(e * KNB + tid) * 2 + 1];
  __syncthreads();

  {  // gather k/v rows (float4), f16 copies, per-head q.k dots for free
    const int j = tid >> 3;   // pair within segment (0..15)
    const int cg = tid & 7;   // head (cols cg*16 .. cg*16+15)
    const int d = sdst[j];
    const float4* k4 = (const float4*)(kf + (size_t)d * HIDDEN + cg * 16);
    const float4* v4 = (const float4*)(vf + (size_t)d * HIDDEN + cg * 16);
    const float4* q4 = (const float4*)(sqrow + cg * 16);
    float4* svf4 = (float4*)(svf + j * 128 + cg * 16);
    float dotv = 0.0f;
#pragma unroll
    for (int i = 0; i < 2; ++i) {
      const float4 ka = k4[2 * i], kb = k4[2 * i + 1];
      const float4 va = v4[2 * i], vb = v4[2 * i + 1];
      const float4 qa = q4[2 * i], qb = q4[2 * i + 1];
      svf4[2 * i] = va; svf4[2 * i + 1] = vb;
      dotv += qa.x * ka.x + qa.y * ka.y + qa.z * ka.z + qa.w * ka.w
            + qb.x * kb.x + qb.y * kb.y + qb.z * kb.z + qb.w * kb.w;
      v8h hk, hv;
      hk[0] = (_Float16)ka.x; hk[1] = (_Float16)ka.y;
      hk[2] = (_Float16)ka.z; hk[3] = (_Float16)ka.w;
      hk[4] = (_Float16)kb.x; hk[5] = (_Float16)kb.y;
      hk[6] = (_Float16)kb.z; hk[7] = (_Float16)kb.w;
      hv[0] = (_Float16)va.x; hv[1] = (_Float16)va.y;
      hv[2] = (_Float16)va.z; hv[3] = (_Float16)va.w;
      hv[4] = (_Float16)vb.x; hv[5] = (_Float16)vb.y;
      hv[6] = (_Float16)vb.z; hv[7] = (_Float16)vb.w;
      *(v8h*)(skh + j * SKV_LD + cg * 16 + 8 * i) = hk;
      *(v8h*)(svh + j * SKV_LD + cg * 16 + 8 * i) = hv;
    }
    sdot[j * 8 + cg] = dotv;
  }
  if (tid < 16) {  // coord diff + radial basis
    const int j = tid;
    const int d = sdst[j];
    const float dx = coords[e * 3 + 0] - coords[d * 3 + 0];
    const float dy = coords[e * 3 + 1] - coords[d * 3 + 1];
    const float dz = coords[e * 3 + 2] - coords[d * 3 + 2];
    scd[j * 3 + 0] = dx; scd[j * 3 + 1] = dy; scd[j * 3 + 2] = dz;
    const float dist = sqrtf(dx * dx + dy * dy + dz * dz);
    const float step = CUTOFF / (float)(NRAD - 1);
    const float inv_w = (float)NRAD / CUTOFF;
#pragma unroll
    for (int c = 0; c < NRAD; ++c) {
      const float t = (dist - (float)c * step) * inv_w;
      srb[j * SRB_LD + c] = (_Float16)__expf(-t * t);
    }
  }
  __syncthreads();

  // ---- Phase B: per-head MLP + softmax + aggregation --------------------
  const int m = lane & 15;
  const int ncol = lane & 15;
  const int rbase = (lane < 16) ? 0 : 8;
  const int ko = (lane >> 4) << 3;
  _Float16* myact = sact + wave * 16 * ACT_LD;

  for (int hh = 0; hh < 2; ++hh) {
    const int h = wave * 2 + hh;

    // A fragments of X = [qp(16), kp(16), rb(64), dot(1), 0-pad]
    v16h ax[4];
#pragma unroll
    for (int f = 0; f < 4; ++f) {
      const int c0 = f * 32 + ko;
      ax[f] = combine8(x_run(c0, h, m, sqh, skh, srb, sdot),
                       x_run(c0 + 16, h, m, sqh, skh, srb, sdot));
    }

    // GEMM1 (X @ W1) + SiLU -> activation tile (f16)
#pragma unroll
    for (int nt = 0; nt < 8; ++nt) {
      v8f acc = {};
#pragma unroll
      for (int f = 0; f < 4; ++f)
        acc = wmma_f16(ax[f],
                       load_frag_rm(sW1T + (nt * 16) * W128_LD + f * 32, W128_LD),
                       acc);
      const int col = nt * 16 + ncol;
      const float bias = sb1[col];
#pragma unroll
      for (int r = 0; r < 8; ++r)
        myact[(rbase + r) * ACT_LD + col] = (_Float16)silu_f(acc[r] + bias);
    }
    __syncthreads();

    // GEMM2 (@ W2) + SiLU folded into score = hmid . W3[:,h]
    v16h a2[4];
#pragma unroll
    for (int f = 0; f < 4; ++f) a2[f] = load_frag_rm(myact + f * 32, ACT_LD);
    float sc[8];
#pragma unroll
    for (int r = 0; r < 8; ++r) sc[r] = 0.0f;
#pragma unroll
    for (int nt = 0; nt < 8; ++nt) {
      v8f acc = {};
#pragma unroll
      for (int f = 0; f < 4; ++f)
        acc = wmma_f16(a2[f],
                       load_frag_rm(sW2T + (nt * 16) * W128_LD + f * 32, W128_LD),
                       acc);
      const int col = nt * 16 + ncol;
      const float bias = sb2[col];
      const float w3 = sW3[col * HEADS + h];
#pragma unroll
      for (int r = 0; r < 8; ++r) sc[r] += silu_f(acc[r] + bias) * w3;
    }
#pragma unroll
    for (int r = 0; r < 8; ++r) {  // reduce across the 16-lane column group
#pragma unroll
      for (int msk = 1; msk < 16; msk <<= 1) sc[r] += __shfl_xor(sc[r], msk, 16);
      sc[r] += sb3[h];
    }

    // segment softmax (segment == this 16-row tile)
    float mx = -1e30f;
#pragma unroll
    for (int r = 0; r < 8; ++r) {
      const float o = __shfl_xor(sc[r], 16, 32);
      mx = fmaxf(mx, fmaxf(sc[r], o));
    }
    float se = 0.0f;
    float ew[8];
#pragma unroll
    for (int r = 0; r < 8; ++r) {
      ew[r] = __expf(sc[r] - mx);
      se += ew[r] + __expf(__shfl_xor(sc[r], 16, 32) - mx);
    }
#pragma unroll
    for (int r = 0; r < 8; ++r) sattn[wave * 16 + rbase + r] = ew[r] / se;
    __syncthreads();

    // agg = sum_j attn_j * v_j (head slice, f32)
    if (lane < 16) {
      float agg = 0.0f;
#pragma unroll
      for (int j = 0; j < 16; ++j)
        agg += sattn[wave * 16 + j] * svf[j * 128 + h * HEAD_DIM + lane];
      feats16[(size_t)e * HIDDEN + h * HEAD_DIM + lane] = (_Float16)agg;
    }

    // gate = sigmoid(silu(vp @ Wg1 + bg1) . Wg2 + bg2) (K padded 16->32)
    v8h zlo = *(const v8h*)(svh + m * SKV_LD + h * HEAD_DIM + ko);
    v8h zhi = {};
    const v16h av = combine8(zlo, zhi);
    float gp[8];
#pragma unroll
    for (int r = 0; r < 8; ++r) gp[r] = 0.0f;
#pragma unroll
    for (int nt = 0; nt < 8; ++nt) {
      v8f acc = {};
      acc = wmma_f16(av, load_frag_rm(sWg1T + (nt * 16) * WG_LD, WG_LD), acc);
      const int col = nt * 16 + ncol;
      const float bias = sbg1[col];
      const float w2g = sWg2[col];
#pragma unroll
      for (int r = 0; r < 8; ++r) gp[r] += silu_f(acc[r] + bias) * w2g;
    }
#pragma unroll
    for (int r = 0; r < 8; ++r) {
#pragma unroll
      for (int msk = 1; msk < 16; msk <<= 1) gp[r] += __shfl_xor(gp[r], msk, 16);
      sgate[wave * 16 + rbase + r] = 1.0f / (1.0f + __expf(-(gp[r] + *sbg2v)));
    }
    __syncthreads();

    // coord update: cu += sum_j attn_j * gate_j * coord_diff_j
    if (lane < 3) {
      float cv = 0.0f;
#pragma unroll
      for (int j = 0; j < 16; ++j)
        cv += sattn[wave * 16 + j] * sgate[wave * 16 + j] * scd[j * 3 + lane];
      atomicAdd(&scu[lane], cv);
    }
    __syncthreads();
  }

  if (tid < 3)
    coords_out[e * 3 + tid] = coords[e * 3 + tid] + scu[tid] / (float)HEADS;
}

// ---------------------------------------------------------------------------
// Kernel 3: y = ef + feats @ Wo + bo, then per-row layernorm.
// ---------------------------------------------------------------------------
__global__ __launch_bounds__(128) void out_kernel(
    const _Float16* __restrict__ feats16, const _Float16* __restrict__ WoT,
    const float* __restrict__ ef, const float* __restrict__ bo,
    const float* __restrict__ gamma_, const float* __restrict__ beta_,
    float* __restrict__ yout) {
  const int wave = threadIdx.x >> 5;
  const int tile = blockIdx.x * 4 + wave;
  const int lane = threadIdx.x & 31;
  const int ncol = lane & 15, rbase = (lane < 16) ? 0 : 8;
  v16h a[4];
#pragma unroll
  for (int f = 0; f < 4; ++f)
    a[f] = load_frag_rm(feats16 + (size_t)tile * 16 * HIDDEN + f * 32, HIDDEN);
  v8f acc[8];
  const v8f zero = {};
#pragma unroll
  for (int nt = 0; nt < 8; ++nt) {
    acc[nt] = zero;
#pragma unroll
    for (int f = 0; f < 4; ++f)
      acc[nt] = wmma_f16(a[f], load_frag_rm(WoT + (nt * 16) * HIDDEN + f * 32, HIDDEN),
                         acc[nt]);
    const int col = nt * 16 + ncol;
    const float bias = bo[col];
#pragma unroll
    for (int r = 0; r < 8; ++r) {
      const int row = tile * 16 + rbase + r;
      acc[nt][r] += bias + ef[(size_t)row * HIDDEN + col];
    }
  }
  float s1[8], s2[8];
#pragma unroll
  for (int r = 0; r < 8; ++r) { s1[r] = 0.0f; s2[r] = 0.0f; }
#pragma unroll
  for (int nt = 0; nt < 8; ++nt)
#pragma unroll
    for (int r = 0; r < 8; ++r) {
      const float y = acc[nt][r];
      s1[r] += y; s2[r] += y * y;
    }
#pragma unroll
  for (int r = 0; r < 8; ++r) {
#pragma unroll
    for (int msk = 1; msk < 16; msk <<= 1) {
      s1[r] += __shfl_xor(s1[r], msk, 16);
      s2[r] += __shfl_xor(s2[r], msk, 16);
    }
  }
#pragma unroll
  for (int r = 0; r < 8; ++r) {
    const float mean = s1[r] / (float)HIDDEN;
    const float var = s2[r] / (float)HIDDEN - mean * mean;
    const float inv = rsqrtf(var + LN_EPS);
    const int row = tile * 16 + rbase + r;
#pragma unroll
    for (int nt = 0; nt < 8; ++nt) {
      const int col = nt * 16 + ncol;
      yout[(size_t)row * HIDDEN + col] =
          (acc[nt][r] - mean) * inv * gamma_[col] + beta_[col];
    }
  }
}

// ---------------------------------------------------------------------------
extern "C" void kernel_launch(void* const* d_in, const int* in_sizes, int n_in,
                              void* d_out, int out_size, void* d_ws,
                              size_t ws_size, hipStream_t stream) {
  (void)in_sizes; (void)n_in; (void)out_size; (void)ws_size;
  const float* ef     = (const float*)d_in[0];
  const float* coords = (const float*)d_in[1];
  const float* Wq  = (const float*)d_in[2];
  const float* Wk  = (const float*)d_in[3];
  const float* Wv  = (const float*)d_in[4];
  const float* W1  = (const float*)d_in[5];
  const float* b1  = (const float*)d_in[6];
  const float* W2  = (const float*)d_in[7];
  const float* b2  = (const float*)d_in[8];
  const float* W3  = (const float*)d_in[9];
  const float* b3  = (const float*)d_in[10];
  const float* Wg1 = (const float*)d_in[11];
  const float* bg1 = (const float*)d_in[12];
  const float* Wg2 = (const float*)d_in[13];
  const float* bg2 = (const float*)d_in[14];
  const float* Wo  = (const float*)d_in[15];
  const float* bo  = (const float*)d_in[16];
  const float* gm  = (const float*)d_in[17];
  const float* bt  = (const float*)d_in[18];
  const int* pairs = (const int*)d_in[19];

  char* ws = (char*)d_ws;
  size_t off = 0;
  auto carve = [&](size_t bytes) -> char* {
    char* r = ws + off;
    off += (bytes + 255) & ~(size_t)255;
    return r;
  };
  float* qf = (float*)carve((size_t)E_N * HIDDEN * 4);
  float* kf = (float*)carve((size_t)E_N * HIDDEN * 4);
  float* vf = (float*)carve((size_t)E_N * HIDDEN * 4);
  _Float16* ef16    = (_Float16*)carve((size_t)E_N * HIDDEN * 2);
  _Float16* feats16 = (_Float16*)carve((size_t)E_N * HIDDEN * 2);
  _Float16* WqT  = (_Float16*)carve(HIDDEN * HIDDEN * 2);
  _Float16* WkT  = (_Float16*)carve(HIDDEN * HIDDEN * 2);
  _Float16* WvT  = (_Float16*)carve(HIDDEN * HIDDEN * 2);
  _Float16* W1T  = (_Float16*)carve(HIDDEN * HIDDEN * 2);
  _Float16* W2T  = (_Float16*)carve(HIDDEN * HIDDEN * 2);
  _Float16* Wg1T = (_Float16*)carve(HIDDEN * 32 * 2);
  _Float16* WoT  = (_Float16*)carve(HIDDEN * HIDDEN * 2);

  float* yout = (float*)d_out;
  float* coords_out = (float*)d_out + (size_t)E_N * HIDDEN;

  prep_kernel<<<512, 256, 0, stream>>>(ef, Wq, Wk, Wv, W1, W2, Wg1, Wo, ef16,
                                       WqT, WkT, WvT, W1T, W2T, Wg1T, WoT);
  qkv_kernel<<<dim3(E_N / 64, 3), 128, 0, stream>>>(ef16, WqT, WkT, WvT, qf, kf, vf);
  edge_attn_kernel<<<E_N, 128, K2_LDS_BYTES, stream>>>(
      qf, kf, vf, coords, pairs, W1T, W2T, Wg1T, b1, b2, W3, b3, bg1, Wg2, bg2,
      feats16, coords_out);
  out_kernel<<<E_N / 64, 128, 0, stream>>>(feats16, WoT, ef, bo, gm, bt, yout);
}
